// OrdinaryAttention_49744311222621
// MI455X (gfx1250) — compile-verified, matrix operand order checked by reading, not run on hardware
//
#include <hip/hip_runtime.h>
#include <hip/hip_bf16.h>

typedef __attribute__((ext_vector_type(16))) __bf16 v16bf;
typedef __attribute__((ext_vector_type(8)))  __bf16 v8bf;
typedef __attribute__((ext_vector_type(8)))  float  v8f;

#define HN     16
#define SEQ    2048
#define DMODEL 1024
#define ADIM   64
#define BATCH  4
#define MROWS  (BATCH * SEQ)   // 8192

static __device__ __forceinline__ v16bf cat8(v8bf lo, v8bf hi) {
  return __builtin_shufflevector(lo, hi, 0,1,2,3,4,5,6,7,8,9,10,11,12,13,14,15);
}

static __device__ __forceinline__ v8f wmma_bf16(v16bf a, v16bf b, v8f c) {
  // D = A(16x32 bf16) * B(32x16 bf16) + C(16x16 f32)
  return __builtin_amdgcn_wmma_f32_16x16x32_bf16(false, a, false, b, (short)0, c, false, false);
}

// gfx1250 async DMA: global -> LDS, 16B per lane, tracked by ASYNCcnt.
// GVS addressing: mem = SADDR(u64) + VADDR(i32) + IOFFSET. VDST = LDS byte addr.
static __device__ __forceinline__ void async_b128(unsigned lds_byte, unsigned voff_byte,
                                                  const void* base) {
  asm volatile("global_load_async_to_lds_b128 %0, %1, %2"
               :: "v"(lds_byte), "v"(voff_byte), "s"((unsigned long long)base)
               : "memory");
}

static __device__ __forceinline__ void wait_async0() {
#if __has_builtin(__builtin_amdgcn_s_wait_asynccnt)
  __builtin_amdgcn_s_wait_asynccnt(0);
#else
  asm volatile("s_wait_asynccnt 0x0" ::: "memory");
#endif
}

// empty volatile asm consuming the accumulator: volatile asms keep program
// order, so this pins the producing WMMAs *before* the following wait/barrier,
// letting the in-flight async DMA overlap the matrix math.
static __device__ __forceinline__ void pin(v8f& x) {
  asm volatile("" : "+v"(x));
}

// low 32 bits of a generic LDS pointer == LDS byte offset (flat aperture mapping)
static __device__ __forceinline__ unsigned lds_off(const void* p) {
  return (unsigned)(unsigned long long)p;
}

// ---------------------------------------------------------------------------
// fp32 -> bf16 elementwise convert
// ---------------------------------------------------------------------------
__global__ __launch_bounds__(256)
void cvt_f32_bf16_kernel(const float* __restrict__ in, __bf16* __restrict__ out, int n) {
  int i = blockIdx.x * 256 + threadIdx.x;
  if (i < n) out[i] = (__bf16)in[i];
}

// ---------------------------------------------------------------------------
// W[K][N] fp32  ->  Wt[N][K] bf16   (LDS tile transpose, coalesced both ways)
// ---------------------------------------------------------------------------
__global__ __launch_bounds__(256)
void transpose_bf16_kernel(const float* __restrict__ W, __bf16* __restrict__ Wt,
                           int K, int N) {
  __shared__ float tile[32][33];
  const int bx = blockIdx.x * 32;  // along N
  const int by = blockIdx.y * 32;  // along K
  const int tx = threadIdx.x & 31;
  const int ty = threadIdx.x >> 5;  // 0..7
#pragma unroll
  for (int i = 0; i < 4; i++) {
    int k = by + ty + i * 8;
    tile[ty + i * 8][tx] = W[(size_t)k * N + bx + tx];
  }
  __syncthreads();
#pragma unroll
  for (int i = 0; i < 4; i++) {
    int n = bx + ty + i * 8;
    Wt[(size_t)n * K + by + tx] = (__bf16)tile[tx][ty + i * 8];
  }
}

// ---------------------------------------------------------------------------
// Tiled WMMA GEMM with double-buffered async-DMA staging.
// C[M][N] = A[M][K] * Bt[N][K]^T + bias[N]
// block tile 128x128, BK=32, 256 threads = 8 waves (4 along M x 2 along N),
// each wave computes 32x64 = 2x4 tiles of 16x16.
// out mode: 0 -> bf16 [B,H,S,64]   (q / k layout)
//           1 -> bf16 [B,H,64,S]   (v transposed layout)
//           2 -> f32  row-major [M][N]
// ---------------------------------------------------------------------------
#define G_BM 128
#define G_BN 128
#define G_BK 32
#define G_LDA 40   // bf16 elems per LDS row (32 + 8 pad): 80B row, 16B aligned, bank-spread

__global__ __launch_bounds__(256)
void gemm_bf16_kernel(const __bf16* __restrict__ A,   // [M][K]
                      const __bf16* __restrict__ Bt,  // [N][K]
                      const float*  __restrict__ bias,
                      void* __restrict__ out,
                      int M, int N, int K, int mode) {
  __shared__ __align__(16) __bf16 lA[2][G_BM * G_LDA];
  __shared__ __align__(16) __bf16 lB[2][G_BN * G_LDA];

  const int tid  = threadIdx.x;
  const int lane = tid & 31;
  const int wave = tid >> 5;
  const int wm   = wave & 3;   // 4 waves along M (32 rows each)
  const int wn   = wave >> 2;  // 2 waves along N (64 cols each)
  const int m0   = blockIdx.x * G_BM;
  const int n0   = blockIdx.y * G_BN;
  const int lm   = lane & 15;
  const int kb   = (lane >> 4) * 8;   // ISA 16-bit A/B fragment K mapping

  // per-thread DMA chunk coords: chunks c = tid and tid+256; r = c>>2, kc = c&3
  const int cr = tid >> 2;     // 0..63
  const int ck = tid & 3;      // 0..3
  const unsigned ldsA0 = lds_off(&lA[0][0]);
  const unsigned ldsB0 = lds_off(&lB[0][0]);
  const unsigned bufA  = (unsigned)(G_BM * G_LDA * 2);
  const unsigned bufB  = (unsigned)(G_BN * G_LDA * 2);
  const unsigned lA_t0 = ldsA0 + (unsigned)((cr      ) * G_LDA + ck * 8) * 2;
  const unsigned lA_t1 = ldsA0 + (unsigned)((cr + 64 ) * G_LDA + ck * 8) * 2;
  const unsigned lB_t0 = ldsB0 + (unsigned)((cr      ) * G_LDA + ck * 8) * 2;
  const unsigned lB_t1 = ldsB0 + (unsigned)((cr + 64 ) * G_LDA + ck * 8) * 2;
  const unsigned gA0 = ((unsigned)(m0 + cr     ) * K + ck * 8) * 2;
  const unsigned gA1 = ((unsigned)(m0 + cr + 64) * K + ck * 8) * 2;
  const unsigned gB0 = ((unsigned)(n0 + cr     ) * K + ck * 8) * 2;
  const unsigned gB1 = ((unsigned)(n0 + cr + 64) * K + ck * 8) * 2;

#define G_ISSUE(k0, buf) do {                                   \
    unsigned kby = (unsigned)(k0) * 2;                          \
    async_b128(lA_t0 + (buf) * bufA, gA0 + kby, A);             \
    async_b128(lA_t1 + (buf) * bufA, gA1 + kby, A);             \
    async_b128(lB_t0 + (buf) * bufB, gB0 + kby, Bt);            \
    async_b128(lB_t1 + (buf) * bufB, gB1 + kby, Bt);            \
  } while (0)

  v8f acc[2][4];
#pragma unroll
  for (int i = 0; i < 2; i++)
#pragma unroll
    for (int j = 0; j < 4; j++) acc[i][j] = (v8f)(0.0f);

  const int KT = K / G_BK;
  G_ISSUE(0, 0);
  wait_async0();
  __syncthreads();

  for (int kt = 0; kt < KT; kt++) {
    const int cur = kt & 1;
    if (kt + 1 < KT) G_ISSUE((kt + 1) * G_BK, cur ^ 1);  // DMA next tile during compute

    v16bf af[2], bfm[4];
#pragma unroll
    for (int i = 0; i < 2; i++) {
      int r = wm * 32 + i * 16 + lm;
      v8bf lo = *(const v8bf*)(&lA[cur][r * G_LDA + kb]);
      v8bf hi = *(const v8bf*)(&lA[cur][r * G_LDA + 16 + kb]);
      af[i] = cat8(lo, hi);
    }
#pragma unroll
    for (int j = 0; j < 4; j++) {
      int r = wn * 64 + j * 16 + lm;
      v8bf lo = *(const v8bf*)(&lB[cur][r * G_LDA + kb]);
      v8bf hi = *(const v8bf*)(&lB[cur][r * G_LDA + 16 + kb]);
      bfm[j] = cat8(lo, hi);
    }
#pragma unroll
    for (int i = 0; i < 2; i++)
#pragma unroll
      for (int j = 0; j < 4; j++)
        acc[i][j] = wmma_bf16(af[i], bfm[j], acc[i][j]);

    // keep WMMAs ahead of the async wait so DMA overlaps the matrix math
#pragma unroll
    for (int i = 0; i < 2; i++)
#pragma unroll
      for (int j = 0; j < 4; j++) pin(acc[i][j]);

    wait_async0();
    __syncthreads();
  }
#undef G_ISSUE

  // epilogue: D layout -> lane holds col lm, VGPR r holds row r + 8*(lane>=16)
  const int mh = (lane >> 4) * 8;
#pragma unroll
  for (int i = 0; i < 2; i++) {
#pragma unroll
    for (int j = 0; j < 4; j++) {
      int n = n0 + wn * 64 + j * 16 + lm;
      float bv = bias[n];
#pragma unroll
      for (int r = 0; r < 8; r++) {
        int m = m0 + wm * 32 + i * 16 + r + mh;
        float val = acc[i][j][r] + bv;
        if (mode == 0) {          // [B,H,S,64] bf16
          int b = m / SEQ, s = m % SEQ;
          int h = n >> 6, dd = n & 63;
          ((__bf16*)out)[(((size_t)(b * HN + h)) * SEQ + s) * ADIM + dd] = (__bf16)val;
        } else if (mode == 1) {   // [B,H,64,S] bf16 (v transposed)
          int b = m / SEQ, s = m % SEQ;
          int h = n >> 6, dd = n & 63;
          ((__bf16*)out)[(((size_t)(b * HN + h)) * ADIM + dd) * SEQ + s] = (__bf16)val;
        } else {                  // f32 row-major
          ((float*)out)[(size_t)m * N + n] = val;
        }
      }
    }
  }
}

// ---------------------------------------------------------------------------
// Flash-style attention with double-buffered async-DMA KV staging.
// grid = (B*H, S/64). 128 threads = 4 waves, each wave owns 16 query rows.
// ---------------------------------------------------------------------------
#define A_LDK 72  // 64 + 8 pad bf16 elems per LDS row (144B: 16B aligned)

__global__ __launch_bounds__(128)
void attn_kernel(const __bf16* __restrict__ qb,   // [BH][S][64]
                 const __bf16* __restrict__ kbuf, // [BH][S][64]
                 const __bf16* __restrict__ vt,   // [BH][64][S]
                 __bf16* __restrict__ ctx) {      // [B*S][1024]
  const int bh   = blockIdx.x;
  const int qblk = blockIdx.y;
  const int tid  = threadIdx.x;
  const int lane = tid & 31;
  const int wave = tid >> 5;
  const int lm   = lane & 15;
  const int kb   = (lane >> 4) * 8;
  const int mh   = (lane >> 4) * 8;

  __shared__ __align__(16) __bf16 lK[2][64 * A_LDK];
  __shared__ __align__(16) __bf16 lV[2][64 * A_LDK];
  __shared__ __align__(16) __bf16 lP[4][16 * A_LDK];

  // per-thread DMA chunks: 512 16B chunks per tile / 128 threads = 4 each
  const int acr = tid >> 3;   // 0..15 (row group)
  const int acc_ = tid & 7;   // 0..7  (16B column chunk)
  const unsigned ldsK0 = lds_off(&lK[0][0]);
  const unsigned ldsV0 = lds_off(&lV[0][0]);
  const unsigned kvbuf = (unsigned)(64 * A_LDK * 2);
  unsigned lK_t[4], lV_t[4], gK[4], gV[4];
#pragma unroll
  for (int i = 0; i < 4; i++) {
    int r = acr + 16 * i;
    lK_t[i] = ldsK0 + (unsigned)(r * A_LDK + acc_ * 8) * 2;
    lV_t[i] = ldsV0 + (unsigned)(r * A_LDK + acc_ * 8) * 2;
    gK[i] = (((unsigned)bh * SEQ + r) * ADIM + acc_ * 8) * 2;  // + kv0*ADIM*2
    gV[i] = (((unsigned)bh * ADIM + r) * SEQ + acc_ * 8) * 2;  // + kv0*2
  }

#define KV_ISSUE(kv0, buf) do {                                        \
    unsigned ok = (unsigned)(kv0) * ADIM * 2, ov = (unsigned)(kv0) * 2;\
    _Pragma("unroll")                                                  \
    for (int i = 0; i < 4; i++) {                                      \
      async_b128(lK_t[i] + (buf) * kvbuf, gK[i] + ok, kbuf);           \
      async_b128(lV_t[i] + (buf) * kvbuf, gV[i] + ov, vt);             \
    }                                                                  \
  } while (0)

  // q fragments stay resident (K = d = 64 -> 2 WMMA k-steps)
  v16bf qf[2];
  {
    const __bf16* qrow = qb + ((size_t)bh * SEQ + qblk * 64 + wave * 16 + lm) * ADIM;
#pragma unroll
    for (int ks = 0; ks < 2; ks++) {
      v8bf lo = *(const v8bf*)(qrow + ks * 32 + kb);
      v8bf hi = *(const v8bf*)(qrow + ks * 32 + 16 + kb);
      qf[ks] = cat8(lo, hi);
    }
  }

  float row_max[8], row_sum[8];
  v8f o[4];
#pragma unroll
  for (int r = 0; r < 8; r++) { row_max[r] = -3.0e38f; row_sum[r] = 0.0f; }
#pragma unroll
  for (int dt = 0; dt < 4; dt++) o[dt] = (v8f)(0.0f);

  KV_ISSUE(0, 0);
  wait_async0();
  __syncthreads();

  for (int kvb = 0; kvb < SEQ / 64; kvb++) {
    const int cur = kvb & 1;
    if (kvb + 1 < SEQ / 64) KV_ISSUE((kvb + 1) * 64, cur ^ 1);  // DMA next KV block

    // scores = q @ k^T  (4 column tiles x 2 k-steps)
    v8f sc[4];
#pragma unroll
    for (int tn = 0; tn < 4; tn++) sc[tn] = (v8f)(0.0f);
#pragma unroll
    for (int ks = 0; ks < 2; ks++) {
#pragma unroll
      for (int tn = 0; tn < 4; tn++) {
        int r = tn * 16 + lm;
        v8bf lo = *(const v8bf*)(&lK[cur][r * A_LDK + ks * 32 + kb]);
        v8bf hi = *(const v8bf*)(&lK[cur][r * A_LDK + ks * 32 + 16 + kb]);
        sc[tn] = wmma_bf16(qf[ks], cat8(lo, hi), sc[tn]);
      }
    }

    // online softmax (row stats live across 16-lane halves, wave32)
    const float scale = 0.125f;  // 1/sqrt(64)
#pragma unroll
    for (int r = 0; r < 8; r++) {
      float mx = -3.0e38f;
#pragma unroll
      for (int tn = 0; tn < 4; tn++) {
        sc[tn][r] *= scale;
        mx = fmaxf(mx, sc[tn][r]);
      }
#pragma unroll
      for (int off = 8; off >= 1; off >>= 1)
        mx = fmaxf(mx, __shfl_xor(mx, off, 16));
      float newm  = fmaxf(row_max[r], mx);
      float alpha = __expf(row_max[r] - newm);
      row_max[r] = newm;
      float psum = 0.0f;
#pragma unroll
      for (int tn = 0; tn < 4; tn++) {
        float p = __expf(sc[tn][r] - newm);
        sc[tn][r] = p;
        psum += p;
      }
#pragma unroll
      for (int off = 8; off >= 1; off >>= 1)
        psum += __shfl_xor(psum, off, 16);
      row_sum[r] = row_sum[r] * alpha + psum;
#pragma unroll
      for (int dt = 0; dt < 4; dt++) o[dt][r] *= alpha;
      // park P in per-wave LDS to re-enter as A-fragments (intra-wave DS order)
      int rl = r + mh;
#pragma unroll
      for (int tn = 0; tn < 4; tn++)
        lP[wave][rl * A_LDK + tn * 16 + lm] = (__bf16)sc[tn][r];
    }

    // context += P @ V
#pragma unroll
    for (int ks = 0; ks < 2; ks++) {
      v8bf plo = *(const v8bf*)(&lP[wave][lm * A_LDK + ks * 32 + kb]);
      v8bf phi = *(const v8bf*)(&lP[wave][lm * A_LDK + ks * 32 + 16 + kb]);
      v16bf pf = cat8(plo, phi);
#pragma unroll
      for (int dt = 0; dt < 4; dt++) {
        int r = dt * 16 + lm;
        v8bf lo = *(const v8bf*)(&lV[cur][r * A_LDK + ks * 32 + kb]);
        v8bf hi = *(const v8bf*)(&lV[cur][r * A_LDK + ks * 32 + 16 + kb]);
        o[dt] = wmma_bf16(pf, cat8(lo, hi), o[dt]);
      }
    }

    // keep PV WMMAs ahead of the async wait so KV DMA overlaps the matrix math
#pragma unroll
    for (int dt = 0; dt < 4; dt++) pin(o[dt]);

    wait_async0();
    __syncthreads();
  }
#undef KV_ISSUE

  // finalize: ctx[(b*S+s)][h*64 + d] bf16
  const int b = bh >> 4, h = bh & 15;
#pragma unroll
  for (int r = 0; r < 8; r++) {
    float inv = 1.0f / row_sum[r];
    int s = qblk * 64 + wave * 16 + r + mh;
    size_t base = ((size_t)(b * SEQ + s)) * DMODEL + h * ADIM;
#pragma unroll
    for (int dt = 0; dt < 4; dt++)
      ctx[base + dt * 16 + lm] = (__bf16)(o[dt][r] * inv);
  }
}

// ---------------------------------------------------------------------------
// residual add + LayerNorm, one 256-thread block per row of 1024
// ---------------------------------------------------------------------------
__global__ __launch_bounds__(256)
void ln_kernel(const float* __restrict__ proj, const float* __restrict__ resid,
               const float* __restrict__ g, const float* __restrict__ b,
               float* __restrict__ out) {
  const int row = blockIdx.x;
  const int tid = threadIdx.x;
  __shared__ float ssum[8], ssq[8];
  float x[4];
  float s = 0.0f, sq = 0.0f;
#pragma unroll
  for (int i = 0; i < 4; i++) {
    int c = tid + i * 256;
    float v = proj[(size_t)row * DMODEL + c] + resid[(size_t)row * DMODEL + c];
    x[i] = v; s += v; sq += v * v;
  }
#pragma unroll
  for (int off = 16; off >= 1; off >>= 1) {
    s  += __shfl_xor(s,  off, 32);
    sq += __shfl_xor(sq, off, 32);
  }
  if ((tid & 31) == 0) { ssum[tid >> 5] = s; ssq[tid >> 5] = sq; }
  __syncthreads();
  s = 0.0f; sq = 0.0f;
#pragma unroll
  for (int w = 0; w < 8; w++) { s += ssum[w]; sq += ssq[w]; }
  float mu  = s * (1.0f / DMODEL);
  float var = sq * (1.0f / DMODEL) - mu * mu;
  float rs  = rsqrtf(var + 1e-5f);
#pragma unroll
  for (int i = 0; i < 4; i++) {
    int c = tid + i * 256;
    out[(size_t)row * DMODEL + c] = (x[i] - mu) * rs * g[c] + b[c];
  }
}

// ---------------------------------------------------------------------------
extern "C" void kernel_launch(void* const* d_in, const int* in_sizes, int n_in,
                              void* d_out, int out_size, void* d_ws, size_t ws_size,
                              hipStream_t stream) {
  (void)in_sizes; (void)n_in; (void)out_size; (void)ws_size;
  const float* Q    = (const float*)d_in[0];
  const float* Wq   = (const float*)d_in[1];
  const float* bq   = (const float*)d_in[2];
  const float* Wk   = (const float*)d_in[3];
  const float* bk   = (const float*)d_in[4];
  const float* Wv   = (const float*)d_in[5];
  const float* bv   = (const float*)d_in[6];
  const float* Wo   = (const float*)d_in[7];
  const float* bo   = (const float*)d_in[8];
  const float* ln_g = (const float*)d_in[9];
  const float* ln_b = (const float*)d_in[10];

  const size_t ND = (size_t)MROWS * DMODEL;     // 8192*1024
  const size_t WW = (size_t)DMODEL * DMODEL;    // 1024*1024

  char* ws = (char*)d_ws;
  size_t off = 0;
  __bf16* Xbf = (__bf16*)(ws + off); off += ND * 2;
  __bf16* WqT = (__bf16*)(ws + off); off += WW * 2;
  __bf16* WkT = (__bf16*)(ws + off); off += WW * 2;
  __bf16* WvT = (__bf16*)(ws + off); off += WW * 2;
  __bf16* WoT = (__bf16*)(ws + off); off += WW * 2;
  __bf16* qbf = (__bf16*)(ws + off); off += ND * 2;   // [B,H,S,64]
  __bf16* kbf = (__bf16*)(ws + off); off += ND * 2;   // [B,H,S,64]
  __bf16* vtb = (__bf16*)(ws + off); off += ND * 2;   // [B,H,64,S]
  __bf16* ctx = (__bf16*)(ws + off); off += ND * 2;   // [B*S, 1024]
  float*  prj = (float*)(ws + off);  off += ND * 4;   // [B*S, 1024]

  // 1) precision staging
  cvt_f32_bf16_kernel<<<(int)((ND + 255) / 256), 256, 0, stream>>>(Q, Xbf, (int)ND);
  dim3 tg(DMODEL / 32, DMODEL / 32);
  transpose_bf16_kernel<<<tg, 256, 0, stream>>>(Wq, WqT, DMODEL, DMODEL);
  transpose_bf16_kernel<<<tg, 256, 0, stream>>>(Wk, WkT, DMODEL, DMODEL);
  transpose_bf16_kernel<<<tg, 256, 0, stream>>>(Wv, WvT, DMODEL, DMODEL);
  transpose_bf16_kernel<<<tg, 256, 0, stream>>>(Wo, WoT, DMODEL, DMODEL);

  // 2) QKV projections (WMMA GEMMs scattering into attention layouts)
  dim3 gg(MROWS / G_BM, DMODEL / G_BN);
  gemm_bf16_kernel<<<gg, 256, 0, stream>>>(Xbf, WqT, bq, qbf, MROWS, DMODEL, DMODEL, 0);
  gemm_bf16_kernel<<<gg, 256, 0, stream>>>(Xbf, WkT, bk, kbf, MROWS, DMODEL, DMODEL, 0);
  gemm_bf16_kernel<<<gg, 256, 0, stream>>>(Xbf, WvT, bv, vtb, MROWS, DMODEL, DMODEL, 1);

  // 3) flash attention
  attn_kernel<<<dim3(BATCH * HN, SEQ / 64), 128, 0, stream>>>(qbf, kbf, vtb, ctx);

  // 4) output projection (f32 out)
  gemm_bf16_kernel<<<gg, 256, 0, stream>>>(ctx, WoT, bo, prj, MROWS, DMODEL, DMODEL, 2);

  // 5) residual + layernorm
  ln_kernel<<<MROWS, 256, 0, stream>>>(prj, Q, ln_g, ln_b, (float*)d_out);
}
